// RNN_25409026523753
// MI455X (gfx1250) — compile-verified, matrix operand order checked by reading, not run on hardware
//
#include <hip/hip_runtime.h>

// ---------------------------------------------------------------------------
// LSTM encoder/decoder, persistent WMMA kernel for gfx1250 (MI455X).
//   N=512 batch, H=512 hidden, D=9, 257 encode + 256 decode steps.
//   Grid: 128 persistent blocks = 32 row-groups (16 batch rows) x 4 gate-split
//   blocks (128 hidden units each). f16 WMMA 16x16x32, f32 accumulate.
//   Software-pipelined fragment loads; async global->LDS h staging.
// ---------------------------------------------------------------------------

typedef __attribute__((ext_vector_type(16))) _Float16 v16h;
typedef __attribute__((ext_vector_type(8)))  float    v8f;
typedef __attribute__((ext_vector_type(4)))  float    f32x4;
typedef __attribute__((ext_vector_type(4)))  int      v4i;

union FragH {            // one WMMA A/B operand: 16 f16 per lane = 32 bytes
  v16h  v;
  f32x4 f4[2];
};

#if __has_builtin(__builtin_amdgcn_global_load_async_to_lds_b128) && \
    __has_builtin(__builtin_amdgcn_s_wait_asynccnt)
#define USE_ASYNC_H 1
typedef __attribute__((address_space(1))) v4i gv4i;   // global int4
typedef __attribute__((address_space(3))) v4i lv4i;   // LDS int4
#endif

#define NH      512
#define ND      9
#define NT      256
#define NSTEPS  513            // 257 encode + 256 decode updates
#define ROWS    16             // batch rows per group
#define NGRP    32
#define GSPLIT  4              // gate-split blocks per group
#define NBLK    (NGRP * GSPLIT)
#define UNITS   128            // hidden units per split block
#define GCOLS   512            // gate columns per block (4 gates x 128 units)

// workspace layout (bytes)
#define WSW_BYTES  (4u * 32u * 16u * 32u * 32u)     // 2 MiB swizzled f16 W_hh
#define HBUF_BYTES (2u * NGRP * ROWS * NH * 2u)     // 1 MiB ping-pong f16 h
#define WSW_OFF    0u
#define HBUF_OFF   (WSW_OFF + WSW_BYTES)
#define BAR_OFF    (HBUF_OFF + HBUF_BYTES)
#define HBUF_STRIDE (NGRP * ROWS * NH * 2u)         // bytes per ping-pong buf

// dynamic LDS carve (bytes)
#define L_H     0                       // ROWS*NH f16            = 16384
#define L_GATES (L_H + 16384)           // ROWS*GCOLS f32         = 32768
#define L_C     (L_GATES + 32768)       // ROWS*UNITS f32         = 8192
#define L_BIAS  (L_C + 8192)            // GCOLS f32              = 2048
#define L_WIH   (L_BIAS + 2048)         // GCOLS*9 f32            = 18432
#define L_X     (L_WIH + 18432)         // ROWS*9 f32 (pad 640)
#define L_FCW   (L_X + 640)             // 9*NH f32               = 18432
#define L_FCB   (L_FCW + 18432)         // 9 f32 (pad 64)
#define SMEM_BYTES (L_FCB + 64)         // = 96960 (< 320KB/WGP)

__device__ __forceinline__ float sigm(float x) {
  return 1.0f / (1.0f + __expf(-x));
}

// ---- prep: zero group barriers, broadcast h0 into ping buffer 0 (f16) ----
__global__ void init_state_kernel(const float* __restrict__ h0,
                                  char* __restrict__ ws) {
  int i = blockIdx.x * 256 + threadIdx.x;
  unsigned* bar = (unsigned*)(ws + BAR_OFF);
  _Float16* hb  = (_Float16*)(ws + HBUF_OFF);
  if (i < NGRP) bar[i] = 0u;
  if (i < NGRP * ROWS * NH) hb[i] = (_Float16)h0[i & (NH - 1)];
}

// ---- prep: swizzle W_hh[2048,512] f32 -> per-lane f16 B fragments ----
__global__ void swizzle_whh_kernel(const float* __restrict__ W_hh,
                                   char* __restrict__ ws) {
  int e = blockIdx.x * 256 + threadIdx.x;          // element index, f16 units
  if (e >= (int)(WSW_BYTES / 2u)) return;
  int ve   = e & 15;           // f16 slot within lane's 32B
  int lane = (e >> 4) & 31;
  int kt   = (e >> 9) & 15;
  int ct   = (e >> 13) & 31;
  int b    = (e >> 18) & 3;
  int v    = ve >> 1, hi = ve & 1;
  int half = lane >> 4, nn = lane & 15;
  int k  = kt * 32 + (v >> 2) * 16 + half * 8 + (v & 3) * 2 + hi;
  int gc = ct * 16 + nn;                 // block-local gate column
  int q = gc >> 7, jj = gc & 127;        // gate (i,f,g,o), unit within block
  int g = q * 512 + b * UNITS + jj;      // row in W_hh
  ((_Float16*)(ws + WSW_OFF))[e] = (_Float16)W_hh[g * NH + k];
}

// ---- persistent LSTM kernel ----
__global__ __launch_bounds__(256, 1)
void lstm_persistent_kernel(const float* __restrict__ x,
                            const float* __restrict__ W_ih,
                            const float* __restrict__ b_ih,
                            const float* __restrict__ b_hh,
                            const float* __restrict__ fc_W,
                            const float* __restrict__ fc_b,
                            const float* __restrict__ c0,
                            char* __restrict__ ws,
                            float* __restrict__ out) {
  extern __shared__ char smem[];
  _Float16* h_lds   = (_Float16*)(smem + L_H);
  float*    gates   = (float*)(smem + L_GATES);
  float*    c_lds   = (float*)(smem + L_C);
  float*    bias    = (float*)(smem + L_BIAS);
  float*    wih     = (float*)(smem + L_WIH);
  float*    x_lds   = (float*)(smem + L_X);
  float*    fcw     = (float*)(smem + L_FCW);
  float*    fcb     = (float*)(smem + L_FCB);

  const int tid = threadIdx.x;
  const int grp = blockIdx.x >> 2;       // row group (16 batch rows)
  const int b   = blockIdx.x & 3;        // gate-split index (128 units)
  unsigned* bar = (unsigned*)(ws + BAR_OFF) + grp;
  char* hbuf    = ws + HBUF_OFF;
  const char* wswB = ws + WSW_OFF + (size_t)b * (32u * 16u * 32u * 32u);

  // ---- one-time staging ----
  for (int gc = tid; gc < GCOLS; gc += 256) {
    int q = gc >> 7, jj = gc & 127;
    int g = q * 512 + b * UNITS + jj;
    bias[gc] = b_ih[g] + b_hh[g];
    #pragma unroll
    for (int d = 0; d < ND; ++d) wih[gc * ND + d] = W_ih[g * ND + d];
  }
  for (int i = tid; i < ROWS * UNITS; i += 256)
    c_lds[i] = c0[b * UNITS + (i & (UNITS - 1))];
  for (int i = tid; i < ND * NH; i += 256) fcw[i] = fc_W[i];
  if (tid < ND) fcb[tid] = fc_b[tid];
  __syncthreads();

  const int wave = tid >> 5, lane = tid & 31;
  const int half = lane >> 4, nn = lane & 15;

  for (int s = 0; s <= NSTEPS; ++s) {
    // -- stage h_s (16KB f16) from ping-pong global buffer into LDS --
    {
      const char* hsrc = hbuf + (size_t)(s & 1) * HBUF_STRIDE +
                         (size_t)grp * (ROWS * NH * 2u);
#if defined(USE_ASYNC_H)
      #pragma unroll
      for (int j = 0; j < 4; ++j) {
        int off = (tid * 4 + j) * 16;
        __builtin_amdgcn_global_load_async_to_lds_b128(
            (gv4i*)(hsrc + off),
            (lv4i*)((char*)h_lds + off),
            0, 0);
      }
      __builtin_amdgcn_s_wait_asynccnt(0);
#else
      const f32x4* hs4 = (const f32x4*)hsrc;
      f32x4* hdst = (f32x4*)h_lds;
      #pragma unroll
      for (int j = 0; j < 4; ++j) hdst[tid * 4 + j] = hs4[tid * 4 + j];
#endif
    }
    if (s < 257 && tid < ROWS * ND) {
      int m = tid / ND, d = tid % ND;
      x_lds[tid] = x[(size_t)(grp * ROWS + m) * (257 * ND) + s * ND + d];
    }
    __syncthreads();

    // -- fc head: out[t] = sigmoid(h_{s} @ fc_W.T + fc_b), t = s-258 --
    if (b == 0 && s >= 258 && tid < ROWS * ND) {
      int t = s - 258, m = tid / ND, d = tid % ND;
      float sum = fcb[d];
      for (int k = 0; k < NH; ++k)
        sum += (float)h_lds[m * NH + k] * fcw[d * NH + k];
      out[(size_t)(grp * ROWS + m) * (NT * ND) + t * ND + d] = sigm(sum);
    }
    if (s == NSTEPS) break;

    // -- recurrent GEMM: gates[16,512] = h[16,512] @ W_hh_slice.T --
    //    double-buffered software pipeline: load frags for kt+1, then 4 WMMAs
    {
      v8f acc[4] = {};
      FragH aF[2];
      FragH bF[2][4];
      {
        const _Float16* hp = h_lds + (lane & 15) * NH + half * 8;
        aF[0].f4[0] = *(const f32x4*)hp;
        aF[0].f4[1] = *(const f32x4*)(hp + 16);
        #pragma unroll
        for (int ci = 0; ci < 4; ++ci) {
          const char* bp =
              wswB + (size_t)(((wave * 4 + ci) * 16) * 32 + lane) * 32;
          bF[0][ci].f4[0] = *(const f32x4*)bp;
          bF[0][ci].f4[1] = *(const f32x4*)(bp + 16);
        }
      }
      #pragma unroll
      for (int kt = 0; kt < 16; ++kt) {
        const int cur = kt & 1, nxt = cur ^ 1;
        if (kt < 15) {
          const _Float16* hp =
              h_lds + (lane & 15) * NH + (kt + 1) * 32 + half * 8;
          aF[nxt].f4[0] = *(const f32x4*)hp;
          aF[nxt].f4[1] = *(const f32x4*)(hp + 16);
          #pragma unroll
          for (int ci = 0; ci < 4; ++ci) {
            const char* bp =
                wswB +
                (size_t)(((wave * 4 + ci) * 16 + (kt + 1)) * 32 + lane) * 32;
            bF[nxt][ci].f4[0] = *(const f32x4*)bp;
            bF[nxt][ci].f4[1] = *(const f32x4*)(bp + 16);
          }
        }
        #pragma unroll
        for (int ci = 0; ci < 4; ++ci)
          acc[ci] = __builtin_amdgcn_wmma_f32_16x16x32_f16(
              false, aF[cur].v, false, bF[cur][ci].v, (short)0, acc[ci],
              false, false);
      }
      #pragma unroll
      for (int ci = 0; ci < 4; ++ci) {
        int gc0 = (wave * 4 + ci) * 16 + nn;
        #pragma unroll
        for (int r = 0; r < 8; ++r)
          gates[(r + 8 * half) * GCOLS + gc0] = acc[ci][r];
      }
    }
    __syncthreads();

    // -- gate nonlinearity + state update for this block's 128 units --
    {
      _Float16* hdst =
          (_Float16*)(hbuf + (size_t)((s + 1) & 1) * HBUF_STRIDE +
                      (size_t)grp * (ROWS * NH * 2u));
      #pragma unroll
      for (int it = 0; it < 8; ++it) {
        int i = tid + it * 256;               // (m, jj) over 16x128
        int m = i >> 7, jj = i & 127;
        float gv[4];
        #pragma unroll
        for (int q = 0; q < 4; ++q) {
          int gc = q * UNITS + jj;
          float v = gates[m * GCOLS + gc] + bias[gc];
          if (s < 257) {
            #pragma unroll
            for (int d = 0; d < ND; ++d)
              v += x_lds[m * ND + d] * wih[gc * ND + d];
          }
          gv[q] = v;
        }
        float ig = sigm(gv[0]), fg = sigm(gv[1]);
        float gg = tanhf(gv[2]), og = sigm(gv[3]);
        float cn = fg * c_lds[i] + ig * gg;
        c_lds[i] = cn;
        hdst[m * NH + b * UNITS + jj] = (_Float16)(og * tanhf(cn));
      }
    }

    // -- 4-block group barrier (monotonic counter, one barrier/step) --
    __threadfence();
    __syncthreads();
    if (tid == 0) {
      atomicAdd(bar, 1u);
      unsigned target = (unsigned)(s + 1) * GSPLIT;
      while (atomicAdd(bar, 0u) < target) __builtin_amdgcn_s_sleep(2);
    }
    __syncthreads();
    __threadfence();
  }
}

extern "C" void kernel_launch(void* const* d_in, const int* in_sizes, int n_in,
                              void* d_out, int out_size, void* d_ws, size_t ws_size,
                              hipStream_t stream) {
  const float* x    = (const float*)d_in[0];
  const float* W_ih = (const float*)d_in[1];
  const float* W_hh = (const float*)d_in[2];
  const float* b_ih = (const float*)d_in[3];
  const float* b_hh = (const float*)d_in[4];
  const float* fc_W = (const float*)d_in[5];
  const float* fc_b = (const float*)d_in[6];
  const float* h0   = (const float*)d_in[7];
  const float* c0   = (const float*)d_in[8];
  float* out = (float*)d_out;
  char* ws = (char*)d_ws;
  (void)in_sizes; (void)n_in; (void)out_size; (void)ws_size;

  init_state_kernel<<<1024, 256, 0, stream>>>(h0, ws);
  swizzle_whh_kernel<<<(WSW_BYTES / 2u + 255) / 256, 256, 0, stream>>>(W_hh, ws);
  lstm_persistent_kernel<<<NBLK, 256, SMEM_BYTES, stream>>>(
      x, W_ih, b_ih, b_hh, fc_W, fc_b, c0, ws, out);
}
